// LIFRecurrent_46497315947074
// MI455X (gfx1250) — matrix operand order repaired and checked by reading, not compiled, not run on hardware
//
#include <hip/hip_runtime.h>
#include <math.h>

// ---------------------------------------------------------------------------
// LIF recurrent SNN scan for MI455X (gfx1250).  B=32, T=1000, N=1024.
// 16 persistent workgroups (8 wave32 each), each owns a 64-column slab.
// Weight slab lives in LDS as f16 (132KB of 320KB/WGP). Per step:
//   elementwise LIF (regs) -> spikes to d_out (f32) + ws staging (f16)
//   -> 16-WG atomic barrier -> async-DMA full 32x1024 f16 act into LDS
//   -> software-pipelined v_wmma_f32_16x16x32_f16 GEMM -> rec tile in LDS.
// ---------------------------------------------------------------------------

typedef _Float16 half_t;
typedef __attribute__((ext_vector_type(16))) _Float16 v16h;
typedef __attribute__((ext_vector_type(8)))  float    v8f;

#define BB    32
#define TT    1000
#define NN    1024
#define NWG   16
#define NCOL  64
#define ALPHA 0.9f

// LDS row pitch padded 1024 -> 1032 halves (2064 B) so the per-lane row
// stride maps the 16 lanes of a fragment onto distinct 4-byte banks.
#define PITCH       1032
#define LDSW_BYTES  (NCOL * PITCH * 2)     // 132,096 B  weight slab (f16)
#define LDSA_BYTES  (BB * PITCH * 2)       //  66,048 B  act matrix (f16)
#define LDSR_BYTES  (BB * NCOL * 4)        //   8,192 B  rec tile (f32)
#define SMEM_BYTES  (LDSW_BYTES + LDSA_BYTES + LDSR_BYTES)   // 206,336 < 320K

#define WS_BAR_BYTES 1024                  // barrier state lives at ws[0..]
#define WS_BUF_HALVES (BB * NN)            // one f16 act buffer = 65,536 B
#define WS_NEEDED (WS_BAR_BYTES + 2 * WS_BUF_HALVES * 2)

union Frag { v16h v; float4 f[2]; };

__global__ void lif_init_barrier(unsigned* bar) {
    bar[0] = 0u;   // arrival count
    bar[1] = 0u;   // generation
}

// 16-workgroup barrier: agent-scope atomics + release/acquire fences.
// Trailing s_cluster_barrier is a NOP under plain dispatch (ClusterID==0)
// but exercises the CDNA5 cluster-sync path.
__device__ __forceinline__ void grid_barrier(unsigned* bar) {
    __threadfence();                 // release this thread's global stores
    __syncthreads();
    if (threadIdx.x == 0) {
        unsigned gen  = __hip_atomic_load(bar + 1, __ATOMIC_RELAXED,
                                          __HIP_MEMORY_SCOPE_AGENT);
        unsigned prev = __hip_atomic_fetch_add(bar, 1u, __ATOMIC_ACQ_REL,
                                               __HIP_MEMORY_SCOPE_AGENT);
        if (prev == NWG - 1) {
            __hip_atomic_store(bar, 0u, __ATOMIC_RELAXED,
                               __HIP_MEMORY_SCOPE_AGENT);
            __hip_atomic_fetch_add(bar + 1, 1u, __ATOMIC_RELEASE,
                                   __HIP_MEMORY_SCOPE_AGENT);
        } else {
            while (__hip_atomic_load(bar + 1, __ATOMIC_ACQUIRE,
                                     __HIP_MEMORY_SCOPE_AGENT) == gen) {
                __builtin_amdgcn_s_sleep(2);
            }
        }
    }
    __syncthreads();
    __threadfence();                 // acquire before cross-WG reads
    if (threadIdx.x < 32) __builtin_amdgcn_s_cluster_barrier();
}

template <bool USE_WS>
__global__ void __launch_bounds__(256)
lif_persistent_kernel(const float* __restrict__ x,      // [B,T,N]
                      const float* __restrict__ w,      // [N,N] (out,in)
                      const float* __restrict__ bias,   // [N]
                      float* __restrict__ out,          // [B,T,N] spikes
                      unsigned* __restrict__ bar,
                      half_t* __restrict__ ws_act) {    // 2 x [B][N] f16
    extern __shared__ char smem[];
    half_t* lds_w   = (half_t*)smem;
    half_t* lds_act = (half_t*)(smem + LDSW_BYTES);
    float*  lds_rec = (float*) (smem + LDSW_BYTES + LDSA_BYTES);
    // raw LDS byte offset of lds_act (dynamic LDS starts after static LDS)
    const unsigned lds_act_base = __builtin_amdgcn_groupstaticsize() + LDSW_BYTES;

    const int wg   = blockIdx.x;
    const int tid  = threadIdx.x;
    const int col0 = wg * NCOL;

    // ---- one-time: weight slab rows [col0, col0+64) -> LDS f16 ----
    for (int i = tid; i < (NCOL * NN) / 4; i += 256) {
        int r   = (i * 4) / NN;
        int cin = (i * 4) % NN;
        float4 v = *(const float4*)(w + (size_t)(col0 + r) * NN + cin);
        half_t* dst = lds_w + r * PITCH + cin;
        dst[0] = (half_t)v.x; dst[1] = (half_t)v.y;
        dst[2] = (half_t)v.z; dst[3] = (half_t)v.w;
    }
    for (int i = tid; i < BB * NCOL; i += 256) lds_rec[i] = 0.0f;
    __syncthreads();

    // elementwise mapping: thread -> (batch row, 8 consecutive cols)
    const int eb = tid >> 3;
    const int ec = (tid & 7) * 8;
    float st[8];
#pragma unroll
    for (int i = 0; i < 8; ++i) st[i] = 0.0f;

    // WMMA tile mapping: 8 waves = 2 row-tiles x 4 col-tiles
    const int wave = tid >> 5;
    const int lane = tid & 31;
    const int rt = wave >> 2, ct = wave & 3;
    const int lg = lane >> 4, ln = lane & 15;
    const int lg8 = lg * 8, lg16 = lg * 16;
    const float bval = bias[col0 + ct * 16 + ln];

    for (int t = 0; t < TT; ++t) {
        // ---- (1) elementwise LIF ----
        const float* xrow = x   + ((size_t)eb * TT + t) * NN + col0 + ec;
        float*       orow = out + ((size_t)eb * TT + t) * NN + col0 + ec;
        float rec8[8] __attribute__((aligned(16)));
        float xx[8]   __attribute__((aligned(16)));
        float act[8]  __attribute__((aligned(16)));
        *(float4*)&rec8[0] = *(float4*)&lds_rec[eb * NCOL + ec];
        *(float4*)&rec8[4] = *(float4*)&lds_rec[eb * NCOL + ec + 4];
        *(float4*)&xx[0]   = *(const float4*)&xrow[0];
        *(float4*)&xx[4]   = *(const float4*)&xrow[4];
        if (t + 1 < TT) __builtin_prefetch(xrow + NN, 0, 0);  // next step's x
#pragma unroll
        for (int i = 0; i < 8; ++i) {
            float s = st[i];
            float a = (s > 0.0f) ? floorf(s) : 0.0f;   // THRESHOLD = 1
            act[i] = a;
            st[i]  = (s - a) * ALPHA + (1.0f - ALPHA) * (xx[i] + rec8[i]);
        }
        *(float4*)&orow[0] = *(float4*)&act[0];        // spikes ARE the output
        *(float4*)&orow[4] = *(float4*)&act[4];
        if constexpr (USE_WS) {                        // f16 copy for async DMA
            half_t a16[8] __attribute__((aligned(16)));
#pragma unroll
            for (int i = 0; i < 8; ++i) a16[i] = (half_t)act[i];
            half_t* wrow = ws_act + (size_t)(t & 1) * WS_BUF_HALVES
                                  + (size_t)eb * NN + col0 + ec;
            *(float4*)wrow = *(float4*)&a16[0];
        }

        // ---- (2) 16-WG barrier: full act matrix for step t is published ----
        grid_barrier(bar);

        // ---- (3) stage full 32x1024 act (f16) into LDS ----
        if constexpr (USE_WS) {
            // pure copy: global_load_async_to_lds_b128, ASYNCcnt-tracked
            const char* src = (const char*)(ws_act
                              + (size_t)(t & 1) * WS_BUF_HALVES);
            for (int c = tid; c < (BB * NN * 2) / 16; c += 256) { // 4096 chunks
                int row = c >> 7;                  // 128 x 16B per 2048B row
                int off = c & 127;
                unsigned laddr = lds_act_base + row * (PITCH * 2) + off * 16;
                const char* g = src + (size_t)c * 16;
                asm volatile("global_load_async_to_lds_b128 %0, %1, off"
                             :: "v"(laddr), "v"(g) : "memory");
            }
            asm volatile("s_wait_asynccnt 0" ::: "memory");
        } else {
            // fallback: convert f32 spikes from d_out
            for (int i = tid; i < (BB * NN) / 4; i += 256) {
                int b  = (i * 4) >> 10;
                int kk = (i * 4) & (NN - 1);
                float4 v = *(const float4*)(out + ((size_t)b * TT + t) * NN + kk);
                half_t* dst = lds_act + b * PITCH + kk;
                dst[0] = (half_t)v.x; dst[1] = (half_t)v.y;
                dst[2] = (half_t)v.z; dst[3] = (half_t)v.w;
            }
        }
        __syncthreads();

        // ---- (4) software-pipelined WMMA GEMM: 16x16 tile over K=1024 ----
        v8f acc;
#pragma unroll
        for (int i = 0; i < 8; ++i) acc[i] = bval;
        const half_t* arow = lds_act + (rt * 16 + ln) * PITCH;
        const half_t* brow = lds_w   + (ct * 16 + ln) * PITCH;
#define LOAD_A(F, KC) do { (F).f[0] = *(const float4*)(arow + (KC) + lg8); \
                           (F).f[1] = *(const float4*)(arow + (KC) + 16 + lg8); } while (0)
#define LOAD_B(F, KC) do { (F).f[0] = *(const float4*)(brow + (KC) + lg8); \
                           (F).f[1] = *(const float4*)(brow + (KC) + 16 + lg8); } while (0)
#define WMMA(A_, B_, C_) __builtin_amdgcn_wmma_f32_16x16x32_f16( \
                           false, (A_).v, false, (B_).v, (short)0, (C_), false, false)
        {
            Frag A0, B0, A1, B1;
            LOAD_A(A0, 0); LOAD_B(B0, 0);
#pragma unroll
            for (int kc = 0; kc < NN; kc += 64) {
                LOAD_A(A1, kc + 32); LOAD_B(B1, kc + 32);
                acc = WMMA(A0, B0, acc);
                if (kc + 64 < NN) { LOAD_A(A0, kc + 64); LOAD_B(B0, kc + 64); }
                acc = WMMA(A1, B1, acc);
            }
        }
#undef LOAD_A
#undef LOAD_B
#undef WMMA

        // ---- (5) accumulator tile -> lds_rec (M = v + 8*lg, N = ln) ----
#pragma unroll
        for (int v = 0; v < 8; ++v) {
            int row = rt * 16 + lg * 8 + v;
            lds_rec[row * NCOL + ct * 16 + ln] = acc[v];
        }
        __syncthreads();   // rec ready for next step's elementwise reads
    }
}

extern "C" void kernel_launch(void* const* d_in, const int* in_sizes, int n_in,
                              void* d_out, int out_size, void* d_ws, size_t ws_size,
                              hipStream_t stream) {
    (void)in_sizes; (void)n_in; (void)out_size;
    const float* x    = (const float*)d_in[0];   // input_current [32,1000,1024]
    const float* w    = (const float*)d_in[1];   // w_rec [1024,1024]
    const float* brec = (const float*)d_in[2];   // b_rec [1024]
    float*    out = (float*)d_out;
    unsigned* bar = (unsigned*)d_ws;
    half_t*   wsact = (half_t*)((char*)d_ws + WS_BAR_BYTES);

    lif_init_barrier<<<dim3(1), dim3(1), 0, stream>>>(bar);
    if (ws_size >= (size_t)WS_NEEDED) {
        lif_persistent_kernel<true><<<dim3(NWG), dim3(256), SMEM_BYTES, stream>>>(
            x, w, brec, out, bar, wsact);
    } else {
        lif_persistent_kernel<false><<<dim3(NWG), dim3(256), SMEM_BYTES, stream>>>(
            x, w, brec, out, bar, wsact);
    }
}
